// DiT_90993177133383
// MI455X (gfx1250) — matrix operand order repaired and checked by reading, not compiled, not run on hardware
//
#include <hip/hip_runtime.h>
#include <math.h>

// ---------------------------------------------------------------------------
// Types for CDNA5 WMMA (wave32, 16x16x32 bf16 -> f32)
// ---------------------------------------------------------------------------
typedef __bf16 bf16_t;
typedef bf16_t bf16x16 __attribute__((ext_vector_type(16)));
typedef bf16_t bf16x2  __attribute__((ext_vector_type(2)));
typedef float  f32x8   __attribute__((ext_vector_type(8)));

union Frag { bf16x16 v; unsigned int u[8]; };

// native f32 -> bf16 (RNE), lowers to v_cvt_pk_bf16_f32 on gfx1250
static __device__ __forceinline__ unsigned short bfbits(float f) {
  union { bf16_t h; unsigned short u; } c; c.h = (bf16_t)f; return c.u;
}
static __device__ __forceinline__ unsigned int pk2bf(float a, float b) {
  union { bf16x2 h; unsigned int u; } c;
  c.h[0] = (bf16_t)a; c.h[1] = (bf16_t)b;
  return c.u;
}

static __device__ __forceinline__ f32x8 zero8() {
  f32x8 z;
#pragma unroll
  for (int i = 0; i < 8; ++i) z[i] = 0.0f;
  return z;
}

static __device__ __forceinline__ f32x8 wmma_bf16(const Frag& a, const Frag& b, f32x8 c) {
  return __builtin_amdgcn_wmma_f32_16x16x32_bf16(false, a.v, false, b.v,
                                                 (short)0, c, false, false);
}

// CDNA5 async DMA: global -> LDS, 16B per lane, tracked by ASYNCcnt.
// LDS operand = logical LDS byte address (low 32 bits of the generic pointer;
// ISA 10.2: LDS aperture maps addr[31:0]).
static __device__ __forceinline__ void async_ld16(unsigned lds_off, const void* g) {
  asm volatile("global_load_async_to_lds_b128 %0, %1, off"
               :: "v"(lds_off), "v"(g) : "memory");
}

// ---------------------------------------------------------------------------
// Problem constants
// ---------------------------------------------------------------------------
#define C_B   2
#define C_S   1024
#define C_D   768
#define C_H   12
#define C_HD  64
#define C_L   12
#define C_T   128          // text tokens
#define C_NS  2048         // B*S rows
#define C_BH  24           // B*H

// GEMM tiling: block 256 threads = 8 wave32; block tile 128x128, wave tile 32x64
#define BM 128
#define BN 128
#define BK 32
#define LSU 18             // LDS row stride in dwords (32 bf16 + pad, even -> 8B aligned)

// Epilogue modes
#define EPI_NONE 0
#define EPI_GELU 1
#define EPI_SIG  2
#define EPI_ADD  3

// ---------------------------------------------------------------------------
// WMMA GEMM:  out[M,N] = act( A[M,K] @ W[N,K]^T + bias[N] ) (+ res[M,N])
// fp32 inputs converted to bf16 while staging tiles into LDS.
// Global loads for step k are issued before the barrier (overlap prior WMMAs).
// ---------------------------------------------------------------------------
__global__ __launch_bounds__(256) void gemm_wmma_kernel(
    const float* __restrict__ A, const float* __restrict__ W,
    const float* __restrict__ bias, const float* __restrict__ res,
    float* __restrict__ out, int M, int N, int K, int epi)
{
  __shared__ __attribute__((aligned(16))) unsigned int As32[BM * LSU];
  __shared__ __attribute__((aligned(16))) unsigned int Bs32[BN * LSU];

  const int t    = threadIdx.x;
  const int wave = t >> 5;
  const int lane = t & 31;
  const int half = lane >> 4;        // 0: lanes 0-15, 1: lanes 16-31
  const int l15  = lane & 15;
  const int wm   = wave & 3;         // 4 waves along M (32 rows each)
  const int wn   = wave >> 2;        // 2 waves along N (64 cols each)
  const int bM   = blockIdx.y * BM;
  const int bN   = blockIdx.x * BN;

  f32x8 acc[2][4];
#pragma unroll
  for (int ti = 0; ti < 2; ++ti)
#pragma unroll
    for (int tj = 0; tj < 4; ++tj) acc[ti][tj] = zero8();

  const int r8  = t >> 3;            // 0..31 base row (stride 32 over 4 iters)
  const int c4  = t & 7;             // 16B chunk within a 32-float row

  for (int k0 = 0; k0 < K; k0 += BK) {
    // ---- issue all global loads first (no LDS access -> overlaps prior math)
    float4 ra[4], rw[4];
#pragma unroll
    for (int i = 0; i < 4; ++i) {
      int r = r8 + i * 32;
      ra[i] = *(const float4*)&A[(size_t)(bM + r) * K + k0 + c4 * 4];
      rw[i] = *(const float4*)&W[(size_t)(bN + r) * K + k0 + c4 * 4];
    }
    __syncthreads();                 // previous iteration's readers are done
#pragma unroll
    for (int i = 0; i < 4; ++i) {
      int r = r8 + i * 32;
      As32[r * LSU + c4 * 2 + 0] = pk2bf(ra[i].x, ra[i].y);
      As32[r * LSU + c4 * 2 + 1] = pk2bf(ra[i].z, ra[i].w);
      Bs32[r * LSU + c4 * 2 + 0] = pk2bf(rw[i].x, rw[i].y);
      Bs32[r * LSU + c4 * 2 + 1] = pk2bf(rw[i].z, rw[i].w);
    }
    __syncthreads();

    Frag fa[2], fb[4];
#pragma unroll
    for (int ti = 0; ti < 2; ++ti) {
      int row = wm * 32 + ti * 16 + l15;
#pragma unroll
      for (int j = 0; j < 8; ++j) {
        // ISA A 16x32 layout (dword units): k/2 = (j&4?8:0)+(j&3)+(half?4:0)
        int u = ((j & 4) ? 8 : 0) + (j & 3) + (half ? 4 : 0);
        fa[ti].u[j] = As32[row * LSU + u];
      }
    }
#pragma unroll
    for (int tj = 0; tj < 4; ++tj) {
      int col = wn * 64 + tj * 16 + l15;
#pragma unroll
      for (int j = 0; j < 8; ++j) {
        int u = ((j & 4) ? 8 : 0) + (j & 3) + (half ? 4 : 0);   // B 32x16 layout (dword units)
        fb[tj].u[j] = Bs32[col * LSU + u];
      }
    }
#pragma unroll
    for (int ti = 0; ti < 2; ++ti)
#pragma unroll
      for (int tj = 0; tj < 4; ++tj)
        acc[ti][tj] = wmma_bf16(fa[ti], fb[tj], acc[ti][tj]);
  }

  // epilogue — C layout: row = vgpr + 8*lanehalf, col = lane&15
#pragma unroll
  for (int ti = 0; ti < 2; ++ti) {
#pragma unroll
    for (int tj = 0; tj < 4; ++tj) {
#pragma unroll
      for (int r = 0; r < 8; ++r) {
        int m = bM + wm * 32 + ti * 16 + r + half * 8;
        int n = bN + wn * 64 + tj * 16 + l15;
        float v = acc[ti][tj][r] + bias[n];
        if (epi == EPI_GELU) {
          v = 0.5f * v * (1.0f + erff(v * 0.70710678118654752f));
        } else if (epi == EPI_SIG) {
          v = 1.0f / (1.0f + __expf(-v));
        } else if (epi == EPI_ADD) {
          v += res[(size_t)m * N + n];
        }
        out[(size_t)m * N + n] = v;
      }
    }
  }
}

// ---------------------------------------------------------------------------
// Flash attention with WMMA on pre-converted bf16 tensors.
//   Q:[BH,Sq,64] bf16   K:[BH,Skv,64] bf16   Vt:[BH,64,Skv] bf16 (pre-transposed)
//   O:[BH,Sq,64] f32
// block = 128 threads (4 waves); block covers 64 q rows; each wave owns 16.
// K/V tiles streamed with GLOBAL_LOAD_ASYNC_TO_LDS_B128, double-buffered:
// tile n+1's DMA overlaps tile n's WMMA work. ASYNCcnt discipline:
// 4 issues/thread/tile -> s_wait_asynccnt 4 retires tile n with n+1 in flight.
// ---------------------------------------------------------------------------
#define QSTR 36   // dwords per LDS row (32 data + 4 pad; 144B, 16B aligned)
#define VSTR 20   // dwords per LDS row (16 data + 4 pad; 80B, 16B aligned)
__global__ __launch_bounds__(128) void flash_wmma_kernel(
    const unsigned short* __restrict__ Q, const unsigned short* __restrict__ Kt,
    const unsigned short* __restrict__ Vt, float* __restrict__ O,
    int Sq, int Skv, float scl)
{
  __shared__ __attribute__((aligned(16))) unsigned int   Qs32[64 * QSTR];
  __shared__ __attribute__((aligned(16))) unsigned int   Ks32[2][32 * QSTR];
  __shared__ __attribute__((aligned(16))) unsigned int   Vs32[2][64 * VSTR];
  __shared__ __attribute__((aligned(16))) unsigned short Ps[4 * 16 * 34];

  const int t    = threadIdx.x;
  const int wave = t >> 5;
  const int lane = t & 31;
  const int half = lane >> 4;
  const int l15  = lane & 15;
  const int bh   = blockIdx.y;
  const int qblk = blockIdx.x * 64;

  const unsigned short* qb = Q  + (size_t)bh * Sq  * 64;
  const unsigned short* kb = Kt + (size_t)bh * Skv * 64;
  const unsigned short* vb = Vt + (size_t)bh * 64 * Skv;
  float*                ob = O  + (size_t)bh * Sq  * 64;

  // stage Q (64 rows x 64 bf16) once: plain 16B copies
#pragma unroll
  for (int i = 0; i < 4; ++i) {
    int idx = t + i * 128;             // 0..511 16B chunks
    int r = idx >> 3, c4 = idx & 7;
    *(uint4*)&Qs32[r * QSTR + c4 * 4] =
        *(const uint4*)&qb[(size_t)(qblk + r) * 64 + c4 * 8];
  }

  // async-DMA one K/V tile (32 keys) into LDS buffer `buf`
  const int kkey = t >> 3, kc4 = t & 7;   // K: 32 keys x 8 chunks, 256 chunks
  const int vd   = t >> 2, vc4 = t & 3;   // V: 64 d-rows x 4 chunks, 256 chunks
  auto issue_tile = [&](int n, int buf) {
    int kv0 = n * 32;
#pragma unroll
    for (int i = 0; i < 2; ++i) {
      int key = kkey + i * 16;
      async_ld16((unsigned)(size_t)&Ks32[buf][key * QSTR + kc4 * 4],
                 &kb[(size_t)(kv0 + key) * 64 + kc4 * 8]);
    }
#pragma unroll
    for (int i = 0; i < 2; ++i) {
      int d = vd + i * 32;
      async_ld16((unsigned)(size_t)&Vs32[buf][d * VSTR + vc4 * 4],
                 &vb[(size_t)d * Skv + kv0 + vc4 * 8]);
    }
  };

  const int qr0 = wave * 16;
  float m_r[8], l_r[8];
  f32x8 o_acc[4];
#pragma unroll
  for (int r = 0; r < 8; ++r) { m_r[r] = -3e38f; l_r[r] = 0.0f; }
#pragma unroll
  for (int nj = 0; nj < 4; ++nj) o_acc[nj] = zero8();

  const int ntiles = Skv >> 5;
  issue_tile(0, 0);

  for (int n = 0; n < ntiles; ++n) {
    const int cur = n & 1;
    if (n + 1 < ntiles) {
      issue_tile(n + 1, (n + 1) & 1);   // overlaps this tile's compute
      asm volatile("s_wait_asynccnt 0x4" ::: "memory");  // tile n landed
    } else {
      asm volatile("s_wait_asynccnt 0x0" ::: "memory");
    }
    __syncthreads();                     // all waves' copies of tile n visible

    const unsigned int* Kst = Ks32[cur];
    const unsigned int* Vst = Vs32[cur];

    // scores S = Q(16x64) @ K^T(64x32): 2 k-steps of 32, 2 key sub-tiles
    f32x8 sc[2]; sc[0] = zero8(); sc[1] = zero8();
#pragma unroll
    for (int ds = 0; ds < 2; ++ds) {
      Frag fq;
      int row = qr0 + l15;
#pragma unroll
      for (int j = 0; j < 8; ++j) {
        int u = ds * 16 + ((j & 4) ? 8 : 0) + (j & 3) + (half ? 4 : 0);
        fq.u[j] = Qs32[row * QSTR + u];
      }
#pragma unroll
      for (int jt = 0; jt < 2; ++jt) {
        Frag fk;
        int key = jt * 16 + l15;
#pragma unroll
        for (int j = 0; j < 8; ++j) {
          int u = ds * 16 + ((j & 4) ? 8 : 0) + (j & 3) + (half ? 4 : 0);
          fk.u[j] = Kst[key * QSTR + u];
        }
        sc[jt] = wmma_bf16(fq, fk, sc[jt]);
      }
    }

    // online softmax per row (row = r + 8*half; 16 lanes hold its 32 cols)
    float p0a[8], p1a[8];
#pragma unroll
    for (int r = 0; r < 8; ++r) {
      float s0 = sc[0][r] * scl;
      float s1 = sc[1][r] * scl;
      float mx = fmaxf(s0, s1);
#pragma unroll
      for (int msk = 1; msk <= 8; msk <<= 1) mx = fmaxf(mx, __shfl_xor(mx, msk));
      float mn = fmaxf(m_r[r], mx);
      float al = __expf(m_r[r] - mn);
      float p0 = __expf(s0 - mn);
      float p1 = __expf(s1 - mn);
      float rs = p0 + p1;
#pragma unroll
      for (int msk = 1; msk <= 8; msk <<= 1) rs += __shfl_xor(rs, msk);
      l_r[r] = l_r[r] * al + rs;
      m_r[r] = mn;
#pragma unroll
      for (int nj = 0; nj < 4; ++nj) o_acc[nj][r] *= al;
      p0a[r] = p0; p1a[r] = p1;
    }

    // bounce P (C-layout) through per-wave LDS into A-fragment layout
    unsigned short* pw = &Ps[wave * 16 * 34];
#pragma unroll
    for (int r = 0; r < 8; ++r) {
      pw[(r + half * 8) * 34 + 0 * 16 + l15] = bfbits(p0a[r]);
      pw[(r + half * 8) * 34 + 1 * 16 + l15] = bfbits(p1a[r]);
    }
    // wave-synchronous: DS ops from the same wave are in-order
    Frag fp;
    {
      int prow = l15;
#pragma unroll
      for (int j = 0; j < 8; ++j) {
        int kk = ((j & 4) ? 16 : 0) + 2 * (j & 3) + (half ? 8 : 0);
        fp.u[j] = *(const unsigned int*)&pw[prow * 34 + kk];
      }
    }
    // O += P(16x32) @ V(32x64)
#pragma unroll
    for (int nj = 0; nj < 4; ++nj) {
      Frag fv;
      int dcol = nj * 16 + l15;
#pragma unroll
      for (int j = 0; j < 8; ++j) {
        int u = ((j & 4) ? 8 : 0) + (j & 3) + (half ? 4 : 0);
        fv.u[j] = Vst[dcol * VSTR + u];
      }
      o_acc[nj] = wmma_bf16(fp, fv, o_acc[nj]);
    }
    __syncthreads();                     // done reading tile n's buffers
  }

#pragma unroll
  for (int nj = 0; nj < 4; ++nj)
#pragma unroll
    for (int r = 0; r < 8; ++r) {
      int row = qblk + qr0 + r + half * 8;
      int d = nj * 16 + l15;
      ob[(size_t)row * 64 + d] = o_acc[nj][r] / l_r[r];
    }
}

// ---------------------------------------------------------------------------
// LayerNorm kernels: one 256-thread block per row of 768
// ---------------------------------------------------------------------------
__device__ __forceinline__ void ln_stats(const float* __restrict__ x, int row,
                                         float v[3], float& mean, float& rstd) {
  __shared__ float s1[256], s2[256];
  int t = threadIdx.x;
  v[0] = x[(size_t)row * C_D + t];
  v[1] = x[(size_t)row * C_D + 256 + t];
  v[2] = x[(size_t)row * C_D + 512 + t];
  s1[t] = v[0] + v[1] + v[2];
  s2[t] = v[0] * v[0] + v[1] * v[1] + v[2] * v[2];
  __syncthreads();
  for (int st = 128; st > 0; st >>= 1) {
    if (t < st) { s1[t] += s1[t + st]; s2[t] += s2[t + st]; }
    __syncthreads();
  }
  mean = s1[0] * (1.0f / 768.0f);
  float var = s2[0] * (1.0f / 768.0f) - mean * mean;
  rstd = rsqrtf(var + 1e-5f);
}

__global__ __launch_bounds__(256) void ln_affine_kernel(
    const float* __restrict__ x, const float* __restrict__ g,
    const float* __restrict__ b, float* __restrict__ out)
{
  int row = blockIdx.x, t = threadIdx.x;
  float v[3], mean, rstd;
  ln_stats(x, row, v, mean, rstd);
#pragma unroll
  for (int i = 0; i < 3; ++i) {
    int d = i * 256 + t;
    out[(size_t)row * C_D + d] = (v[i] - mean) * rstd * g[d] + b[d];
  }
}

__global__ __launch_bounds__(256) void ln_mod_kernel(
    const float* __restrict__ x, const float* __restrict__ scale,
    const float* __restrict__ shift, float* __restrict__ out)
{
  int row = blockIdx.x, t = threadIdx.x;
  int bsel = row >> 10;     // row / S
  float v[3], mean, rstd;
  ln_stats(x, row, v, mean, rstd);
#pragma unroll
  for (int i = 0; i < 3; ++i) {
    int d = i * 256 + t;
    out[(size_t)row * C_D + d] =
        (v[i] - mean) * rstd * scale[bsel * C_D + d] + shift[bsel * C_D + d];
  }
}

// ---------------------------------------------------------------------------
// RoPE + head-transpose, fp32 -> bf16:
//   QKV:[B,S,3D] f32 -> q,k [BH,S,64] bf16, v [BH,64,S] bf16 (transposed)
// ---------------------------------------------------------------------------
__global__ void rope_split_kernel(const float* __restrict__ qkv,
                                  unsigned short* __restrict__ qt,
                                  unsigned short* __restrict__ kt,
                                  unsigned short* __restrict__ vt)
{
  int i = blockIdx.x * blockDim.x + threadIdx.x;
  if (i >= C_B * C_S * C_D) return;
  int d = i & 63;
  int hh = (i >> 6) % C_H;
  int s = (i / C_D) & (C_S - 1);
  int b = i / (C_D * C_S);

  size_t base = (size_t)(b * C_S + s) * (3 * C_D) + hh * 64;
  float qv = qkv[base + d];
  float kv = qkv[base + C_D + d];
  float vv = qkv[base + 2 * C_D + d];

  int j = d & 31;
  float inv = __expf(-(float)(2 * j) * (1.0f / 64.0f) * 9.2103403719761836f); // ln(10000)
  float ang = (float)s * inv;
  float c, sn;
  __sincosf(ang, &sn, &c);
  float qpart, kpart;
  if (d < 32) { qpart = -qkv[base + d + 32]; kpart = -qkv[base + C_D + d + 32]; }
  else        { qpart =  qkv[base + d - 32]; kpart =  qkv[base + C_D + d - 32]; }

  int bh = b * C_H + hh;
  size_t o = ((size_t)bh * C_S + s) * 64 + d;
  qt[o] = bfbits(qv * c + qpart * sn);
  kt[o] = bfbits(kv * c + kpart * sn);
  vt[((size_t)bh * 64 + d) * C_S + s] = bfbits(vv);
}

// q [NS,768] f32 -> [BH,S,64] bf16
__global__ void split_q_kernel(const float* __restrict__ qbuf,
                               unsigned short* __restrict__ qt)
{
  int i = blockIdx.x * blockDim.x + threadIdx.x;
  if (i >= C_B * C_S * C_D) return;
  int d = i & 63;
  int hh = (i >> 6) % C_H;
  int s = (i / C_D) & (C_S - 1);
  int b = i / (C_D * C_S);
  qt[((size_t)(b * C_H + hh) * C_S + s) * 64 + d] = bfbits(qbuf[i]);
}

// kv [B*T,1536] f32 -> k [BH,T,64] bf16, v [BH,64,T] bf16 (transposed)
__global__ void split_kv_kernel(const float* __restrict__ kvbuf,
                                unsigned short* __restrict__ ckt,
                                unsigned short* __restrict__ cvt)
{
  int i = blockIdx.x * blockDim.x + threadIdx.x;
  if (i >= C_B * C_H * C_T * 64) return;
  int d = i & 63;
  int tt = (i >> 6) & (C_T - 1);
  int hh = (i / (64 * C_T)) % C_H;
  int b = i / (64 * C_T * C_H);
  size_t src = (size_t)(b * C_T + tt) * 1536 + hh * 64 + d;
  int bh = b * C_H + hh;
  ckt[i] = bfbits(kvbuf[src]);
  cvt[((size_t)bh * 64 + d) * C_T + tt] = bfbits(kvbuf[src + C_D]);
}

// [BH,S,64] f32 -> [B,S,D] f32, optionally += (residual add)
__global__ void untrans_kernel(const float* __restrict__ att, float* __restrict__ y,
                               int addFlag)
{
  int i = blockIdx.x * blockDim.x + threadIdx.x;
  if (i >= C_B * C_S * C_D) return;
  int d = i & 63;
  int hh = (i >> 6) % C_H;
  int s = (i / C_D) & (C_S - 1);
  int b = i / (C_D * C_S);
  float v = att[((size_t)(b * C_H + hh) * C_S + s) * 64 + d];
  if (addFlag) y[i] += v; else y[i] = v;
}

__global__ void mul_kernel(float* __restrict__ a, const float* __restrict__ g, int n)
{
  int i = blockIdx.x * blockDim.x + threadIdx.x;
  if (i < n) a[i] *= g[i];
}

// ---------------------------------------------------------------------------
// Tiny time-embedding / AdaLN helpers (VALU, negligible cost)
// ---------------------------------------------------------------------------
__global__ void gather_temb_kernel(const float* __restrict__ table,
                                   const int* __restrict__ t, float* __restrict__ out)
{
  int i = blockIdx.x * blockDim.x + threadIdx.x;
  if (i >= C_B * 256) return;
  int b = i >> 8, k = i & 255;
  out[i] = table[(size_t)t[b] * 256 + k];
}

__global__ void small_linear_kernel(const float* __restrict__ in,
                                    const float* __restrict__ W,
                                    const float* __restrict__ bias,
                                    float* __restrict__ out,
                                    int Bn, int K, int Onum, int inSilu, int outSilu)
{
  int i = blockIdx.x * blockDim.x + threadIdx.x;
  if (i >= Bn * Onum) return;
  int b = i / Onum, o = i % Onum;
  float acc = bias[o];
  for (int k = 0; k < K; ++k) {
    float v = in[(size_t)b * K + k];
    if (inSilu) v = v / (1.0f + __expf(-v));
    acc += v * W[(size_t)o * K + k];
  }
  if (outSilu) acc = acc / (1.0f + __expf(-acc));
  out[i] = acc;
}

__global__ void tmean_kernel(const float* __restrict__ text, float* __restrict__ out)
{
  int i = blockIdx.x * blockDim.x + threadIdx.x;
  if (i >= C_B * C_D) return;
  int b = i / C_D, d = i % C_D;
  float s = 0.0f;
  for (int tt = 0; tt < C_T; ++tt) s += text[(size_t)(b * C_T + tt) * C_D + d];
  out[i] = s * (1.0f / (float)C_T);
}

__global__ void modscale_kernel(const float* __restrict__ adat,
                                const float* __restrict__ adax,
                                float* __restrict__ scale, float* __restrict__ shift)
{
  int i = blockIdx.x * blockDim.x + threadIdx.x;
  if (i >= C_B * C_D) return;
  int b = i / C_D, d = i % C_D;
  scale[i] = 1.0f + adat[(size_t)b * 1536 + d] + adax[(size_t)b * 1536 + d];
  shift[i] = adat[(size_t)b * 1536 + C_D + d] + adax[(size_t)b * 1536 + C_D + d];
}

// ---------------------------------------------------------------------------
// Host orchestration
// ---------------------------------------------------------------------------
extern "C" void kernel_launch(void* const* d_in, const int* in_sizes, int n_in,
                              void* d_out, int out_size, void* d_ws, size_t ws_size,
                              hipStream_t stream) {
  (void)in_sizes; (void)n_in; (void)out_size; (void)ws_size;

  const float* x          = (const float*)d_in[0];
  const float* text       = (const float*)d_in[1];
  const int*   t_idx      = (const int*)  d_in[2];
  const float* t_table    = (const float*)d_in[3];
  const float* te_w1      = (const float*)d_in[4];
  const float* te_b1      = (const float*)d_in[5];
  const float* te_w2      = (const float*)d_in[6];
  const float* te_b2      = (const float*)d_in[7];
  const float* ada_time_w = (const float*)d_in[8];
  const float* ada_time_b = (const float*)d_in[9];
  const float* ada_text_w = (const float*)d_in[10];
  const float* ada_text_b = (const float*)d_in[11];
  const float* proj_in_w  = (const float*)d_in[12];
  const float* proj_in_b  = (const float*)d_in[13];
  const float* proj_out_w = (const float*)d_in[14];
  const float* proj_out_b = (const float*)d_in[15];
  const float* n1_g = (const float*)d_in[16];
  const float* n1_b = (const float*)d_in[17];
  const float* qkv_w = (const float*)d_in[18];
  const float* qkv_b = (const float*)d_in[19];
  const float* n2_g = (const float*)d_in[20];
  const float* n2_b = (const float*)d_in[21];
  const float* ca_qkv_w = (const float*)d_in[22];
  const float* ca_qkv_b = (const float*)d_in[23];
  const float* ca_out_w = (const float*)d_in[24];
  const float* ca_out_b = (const float*)d_in[25];
  const float* n3_g = (const float*)d_in[26];
  const float* n3_b = (const float*)d_in[27];
  const float* fc1_w = (const float*)d_in[28];
  const float* fc1_b = (const float*)d_in[29];
  const float* gate_w = (const float*)d_in[30];
  const float* gate_b = (const float*)d_in[31];
  const float* fc2_w = (const float*)d_in[32];
  const float* fc2_b = (const float*)d_in[33];

  float* ws = (float*)d_ws;
  size_t o = 0;
  float* H_   = ws + o; o += (size_t)C_NS * C_D;
  float* XSK  = ws + o; o += (size_t)C_NS * C_D;
  float* XN   = ws + o; o += (size_t)C_NS * C_D;
  float* QKV  = ws + o; o += (size_t)C_NS * 3 * C_D;
  // bf16 tensors (allocated as float-sized regions, used as ushort)
  unsigned short* QT  = (unsigned short*)(ws + o); o += (size_t)C_NS * C_D / 2;
  unsigned short* KT  = (unsigned short*)(ws + o); o += (size_t)C_NS * C_D / 2;
  unsigned short* VT  = (unsigned short*)(ws + o); o += (size_t)C_NS * C_D / 2;
  unsigned short* CKT = (unsigned short*)(ws + o); o += (size_t)C_B * C_H * C_T * 64 / 2;
  unsigned short* CVT = (unsigned short*)(ws + o); o += (size_t)C_B * C_H * C_T * 64 / 2;
  float* ATT  = ws + o; o += (size_t)C_NS * C_D;
  float* CAKV = ws + o; o += (size_t)C_B * C_T * 1536;
  float* CAO  = ws + o; o += (size_t)C_NS * C_D;
  float* UP   = ws + o; o += (size_t)C_NS * 4 * C_D;
  float* GT   = ws + o; o += (size_t)C_NS * 4 * C_D;
  float* TEMB0= ws + o; o += C_B * 256;
  float* TH_  = ws + o; o += C_B * 256;
  float* TEMB = ws + o; o += C_B * 256;
  float* ADAT = ws + o; o += C_B * 1536;
  float* TMEAN= ws + o; o += C_B * C_D;
  float* ADAX = ws + o; o += C_B * 1536;
  float* SCL  = ws + o; o += C_B * C_D;
  float* SHF  = ws + o; o += C_B * C_D;

  const int NE = C_B * C_S * C_D;        // 1572864
  dim3 blk256(256);
  const float attn_scale = 0.125f;       // 1/sqrt(64)

  // ---- time embedding + AdaLN modulation ----
  gather_temb_kernel<<<2, 256, 0, stream>>>(t_table, t_idx, TEMB0);
  small_linear_kernel<<<2, 256, 0, stream>>>(TEMB0, te_w1, te_b1, TH_, C_B, 256, 256, 0, 1);
  small_linear_kernel<<<2, 256, 0, stream>>>(TH_, te_w2, te_b2, TEMB, C_B, 256, 256, 0, 0);
  small_linear_kernel<<<12, 256, 0, stream>>>(TEMB, ada_time_w, ada_time_b, ADAT, C_B, 256, 1536, 1, 0);
  tmean_kernel<<<6, 256, 0, stream>>>(text, TMEAN);
  small_linear_kernel<<<12, 256, 0, stream>>>(TMEAN, ada_text_w, ada_text_b, ADAX, C_B, C_D, 1536, 1, 0);
  modscale_kernel<<<6, 256, 0, stream>>>(ADAT, ADAX, SCL, SHF);

  // ---- x_skip = x @ proj_in^T + b ----
  gemm_wmma_kernel<<<dim3(C_D / BN, C_NS / BM), blk256, 0, stream>>>(
      x, proj_in_w, proj_in_b, nullptr, XSK, C_NS, C_D, C_D, EPI_NONE);

  // ---- h = LN(x) * scale + shift ----
  ln_mod_kernel<<<C_NS, 256, 0, stream>>>(x, SCL, SHF, H_);

  for (int i = 0; i < C_L; ++i) {
    // ===== self-attention with RoPE =====
    ln_affine_kernel<<<C_NS, 256, 0, stream>>>(H_, n1_g + i * C_D, n1_b + i * C_D, XN);
    gemm_wmma_kernel<<<dim3(3 * C_D / BN, C_NS / BM), blk256, 0, stream>>>(
        XN, qkv_w + (size_t)i * 3 * C_D * C_D, qkv_b + (size_t)i * 3 * C_D,
        nullptr, QKV, C_NS, 3 * C_D, C_D, EPI_NONE);
    rope_split_kernel<<<(NE + 255) / 256, blk256, 0, stream>>>(QKV, QT, KT, VT);
    flash_wmma_kernel<<<dim3(C_S / 64, C_BH), dim3(128), 0, stream>>>(
        QT, KT, VT, ATT, C_S, C_S, attn_scale);
    untrans_kernel<<<(NE + 255) / 256, blk256, 0, stream>>>(ATT, H_, 1);  // h += attn

    // ===== cross-attention =====
    ln_affine_kernel<<<C_NS, 256, 0, stream>>>(H_, n2_g + i * C_D, n2_b + i * C_D, XN);
    gemm_wmma_kernel<<<dim3(C_D / BN, C_NS / BM), blk256, 0, stream>>>(
        XN, ca_qkv_w + (size_t)i * 3 * C_D * C_D, ca_qkv_b + (size_t)i * 3 * C_D,
        nullptr, QKV, C_NS, C_D, C_D, EPI_NONE);          // q = first 768 rows of W
    split_q_kernel<<<(NE + 255) / 256, blk256, 0, stream>>>(QKV, QT);
    gemm_wmma_kernel<<<dim3(1536 / BN, (C_B * C_T) / BM), blk256, 0, stream>>>(
        text, ca_qkv_w + (size_t)i * 3 * C_D * C_D + (size_t)C_D * C_D,
        ca_qkv_b + (size_t)i * 3 * C_D + C_D, nullptr, CAKV, C_B * C_T, 1536, C_D, EPI_NONE);
    split_kv_kernel<<<(C_B * C_H * C_T * 64 + 255) / 256, blk256, 0, stream>>>(CAKV, CKT, CVT);
    flash_wmma_kernel<<<dim3(C_S / 64, C_BH), dim3(128), 0, stream>>>(
        QT, CKT, CVT, ATT, C_S, C_T, attn_scale);
    untrans_kernel<<<(NE + 255) / 256, blk256, 0, stream>>>(ATT, CAO, 0);
    gemm_wmma_kernel<<<dim3(C_D / BN, C_NS / BM), blk256, 0, stream>>>(
        CAO, ca_out_w + (size_t)i * C_D * C_D, ca_out_b + (size_t)i * C_D,
        H_, H_, C_NS, C_D, C_D, EPI_ADD);                 // h = caout + h

    // ===== gated MLP =====
    ln_affine_kernel<<<C_NS, 256, 0, stream>>>(H_, n3_g + i * C_D, n3_b + i * C_D, XN);
    gemm_wmma_kernel<<<dim3(4 * C_D / BN, C_NS / BM), blk256, 0, stream>>>(
        XN, fc1_w + (size_t)i * 4 * C_D * C_D, fc1_b + (size_t)i * 4 * C_D,
        nullptr, UP, C_NS, 4 * C_D, C_D, EPI_GELU);
    gemm_wmma_kernel<<<dim3(4 * C_D / BN, C_NS / BM), blk256, 0, stream>>>(
        XN, gate_w + (size_t)i * 4 * C_D * C_D, gate_b + (size_t)i * 4 * C_D,
        nullptr, GT, C_NS, 4 * C_D, C_D, EPI_SIG);
    mul_kernel<<<(C_NS * 4 * C_D + 255) / 256, blk256, 0, stream>>>(UP, GT, C_NS * 4 * C_D);
    gemm_wmma_kernel<<<dim3(C_D / BN, C_NS / BM), blk256, 0, stream>>>(
        UP, fc2_w + (size_t)i * C_D * 4 * C_D, fc2_b + (size_t)i * C_D,
        H_, H_, C_NS, C_D, 4 * C_D, EPI_ADD);             // h = mlp + h
  }

  // ---- out = x_skip + h @ proj_out^T + b ----
  gemm_wmma_kernel<<<dim3(C_D / BN, C_NS / BM), blk256, 0, stream>>>(
      H_, proj_out_w, proj_out_b, XSK, (float*)d_out, C_NS, C_D, C_D, EPI_ADD);
}
